// SALSPL_47863115547249
// MI455X (gfx1250) — matrix operand order repaired and checked by reading, not compiled, not run on hardware
//
#include <hip/hip_runtime.h>
#include <hip/hip_bf16.h>

typedef __attribute__((ext_vector_type(16))) __bf16 v16bf;
typedef __attribute__((ext_vector_type(8)))  __bf16 v8bf;
typedef __attribute__((ext_vector_type(4)))  __bf16 bf16x4;
typedef __attribute__((ext_vector_type(8)))  float  v8f;

#define S_ROWS 2048
#define L_COLS 4096
#define D_DIM  500
#define N_POI  10000
#define N_TIME 48

#define BM 128
#define BN 64
#define BK 32
#define NSTEPS ((D_DIM + BK - 1) / BK)   // 16

// ---------------------------------------------------------------------------
// Kernel 1: bias[l] = dot(id_emb[l,:], socail_uid_emb)
// ---------------------------------------------------------------------------
__global__ __launch_bounds__(256) void bias_kernel(const float* __restrict__ id_emb,
                                                   const float* __restrict__ soc,
                                                   float* __restrict__ bias) {
    __shared__ float red[256];
    const int l = blockIdx.x;
    const int tid = threadIdx.x;
    const float* row = id_emb + (size_t)l * D_DIM;
    float acc = 0.f;
    for (int d = tid; d < D_DIM; d += 256) acc += row[d] * soc[d];
    red[tid] = acc;
    __syncthreads();
    for (int s = 128; s > 0; s >>= 1) {
        if (tid < s) red[tid] += red[tid + s];
        __syncthreads();
    }
    if (tid == 0) bias[l] = red[0];
}

// ---------------------------------------------------------------------------
// Kernel 2: user_energy = w1 * (user_emb @ id_emb^T) + w2 * bias[None,:]
// WMMA bf16 GEMM, block tile 128x64, double-buffered LDS, float4-vectorized
// staging with packed bf16x4 (b64) LDS stores.
//   Alignment: row stride 500 floats; 500 % 4 == 0, k % 4 == 0 -> 16B aligned.
//   Bounds: k multiple of 4 and D=500 -> quad entirely in/out of range.
// ---------------------------------------------------------------------------
__device__ __forceinline__ void stage_tile(int buf, int kb, int tid,
                                           const float* __restrict__ user_emb,
                                           const float* __restrict__ id_emb,
                                           int m0, int n0,
                                           __bf16 (*As)[BM * BK],
                                           __bf16 (*Bs)[BN * BK]) {
    // A tile: 128x32 elems = 1024 quads, 4 per thread
    #pragma unroll
    for (int i = 0; i < (BM * BK) / (4 * 256); ++i) {
        const int qidx = tid + i * 256;
        const int r = qidx >> 3, q = qidx & 7;
        const int k = kb + q * 4;
        float4 f = make_float4(0.f, 0.f, 0.f, 0.f);
        if (k < D_DIM)
            f = *(const float4*)&user_emb[(size_t)(m0 + r) * D_DIM + k];
        bf16x4 h;
        h.x = (__bf16)f.x; h.y = (__bf16)f.y;
        h.z = (__bf16)f.z; h.w = (__bf16)f.w;
        *(bf16x4*)&As[buf][r * BK + q * 4] = h;
    }
    // B tile: 64x32 elems = 512 quads, 2 per thread
    #pragma unroll
    for (int i = 0; i < (BN * BK) / (4 * 256); ++i) {
        const int qidx = tid + i * 256;
        const int r = qidx >> 3, q = qidx & 7;
        const int k = kb + q * 4;
        float4 f = make_float4(0.f, 0.f, 0.f, 0.f);
        if (k < D_DIM)
            f = *(const float4*)&id_emb[(size_t)(n0 + r) * D_DIM + k];
        bf16x4 h;
        h.x = (__bf16)f.x; h.y = (__bf16)f.y;
        h.z = (__bf16)f.z; h.w = (__bf16)f.w;
        *(bf16x4*)&Bs[buf][r * BK + q * 4] = h;
    }
}

__global__ __launch_bounds__(256) void user_energy_wmma(
    const float* __restrict__ user_emb,   // S x D
    const float* __restrict__ id_emb,     // L x D
    const float* __restrict__ bias,       // L
    const float* __restrict__ w1p,
    const float* __restrict__ w2p,
    float* __restrict__ energy)           // S x L (attn_user slice of d_out)
{
    __shared__ __attribute__((aligned(16))) __bf16 As[2][BM * BK];  // 16 KB
    __shared__ __attribute__((aligned(16))) __bf16 Bs[2][BN * BK];  //  8 KB

    const int tid  = threadIdx.x;
    const int wave = tid >> 5;        // 0..7
    const int lane = tid & 31;
    const int half = lane >> 4;       // 0 or 1
    const int l15  = lane & 15;
    const int m0 = blockIdx.y * BM;
    const int n0 = blockIdx.x * BN;

    v8f acc[4] = {};  // 4 N-tiles of 16x16 f32

    stage_tile(0, 0, tid, user_emb, id_emb, m0, n0, As, Bs);
    __syncthreads();

    #pragma unroll            // NSTEPS=16 compile-time: fold addresses + guards
    for (int step = 0; step < NSTEPS; ++step) {
        const int cur = step & 1;
        const int nxt = cur ^ 1;

        // ---- issue ALL fragment loads first (read buffer `cur`) so the
        //      WMMA burst only needs DScnt <= #outstanding staging stores ----
        // A fragment (ISA 16-bit A 16x32 layout):
        //   lanes 0-15:  elems 0..7 = K 0..7,  elems 8..15 = K 16..23
        //   lanes 16-31: elems 0..7 = K 8..15, elems 8..15 = K 24..31
        const int mrow  = wave * 16 + l15;
        const int koffA = half * 8;
        const v8bf alo = *(const v8bf*)&As[cur][mrow * BK + koffA];
        const v8bf ahi = *(const v8bf*)&As[cur][mrow * BK + 16 + koffA];
        const v16bf afrag = __builtin_shufflevector(
            alo, ahi, 0, 1, 2, 3, 4, 5, 6, 7, 8, 9, 10, 11, 12, 13, 14, 15);

        // B fragments (ISA 16-bit B 32x16 layout):
        //   lanes 0-15: elems 0..7 = K 0..7, elems 8..15 = K 16..23; lanes 16-31: K 8..15, K 24..31; N = lane&15
        const int koffB = half * 8;
        v16bf bfrag[4];
        #pragma unroll
        for (int t = 0; t < 4; ++t) {
            const int nrow = t * 16 + l15;
            const v8bf blo = *(const v8bf*)&Bs[cur][nrow * BK + koffB];
            const v8bf bhi = *(const v8bf*)&Bs[cur][nrow * BK + koffB + 16];
            bfrag[t] = __builtin_shufflevector(
                blo, bhi, 0, 1, 2, 3, 4, 5, 6, 7, 8, 9, 10, 11, 12, 13, 14, 15);
        }

        // ---- stage next tile (writes buffer `nxt`; no conflict with reads) ----
        if (step + 1 < NSTEPS)
            stage_tile(nxt, (step + 1) * BK, tid, user_emb, id_emb, m0, n0, As, Bs);

        // ---- WMMA burst ----
        #pragma unroll
        for (int t = 0; t < 4; ++t) {
            acc[t] = __builtin_amdgcn_wmma_f32_16x16x32_bf16(
                false, afrag, false, bfrag[t], (short)0, acc[t], false, false);
        }
        __syncthreads();
    }

    // Epilogue: energy = w1*acc + w2*bias[col]
    const float w1 = *w1p, w2 = *w2p;
    #pragma unroll
    for (int t = 0; t < 4; ++t) {
        const int col = n0 + t * 16 + l15;
        const float bv = w2 * bias[col];
        #pragma unroll
        for (int r = 0; r < 8; ++r) {
            const int row = m0 + wave * 16 + half * 8 + r;   // C/D layout: VGPR r -> M=r(+8)
            energy[(size_t)row * L_COLS + col] = w1 * acc[t][r] + bv;
        }
    }
}

// ---------------------------------------------------------------------------
// Shared softmax-over-LDS-row helper. vals[L_COLS] already populated.
// Final normalize+store is float4-vectorized (rows are 16 KB, 16 B aligned).
// ---------------------------------------------------------------------------
__device__ __forceinline__ void softmax_row_lds(float* vals, float* red, int tid,
                                                float* __restrict__ out_row) {
    float m = -3.402823466e38f;
    for (int l = tid; l < L_COLS; l += 256) m = fmaxf(m, vals[l]);
    red[tid] = m;
    __syncthreads();
    for (int s = 128; s > 0; s >>= 1) {
        if (tid < s) red[tid] = fmaxf(red[tid], red[tid + s]);
        __syncthreads();
    }
    const float mx = red[0];
    __syncthreads();
    float sum = 0.f;
    for (int l = tid; l < L_COLS; l += 256) {
        const float e = __expf(vals[l] - mx);
        vals[l] = e;
        sum += e;
    }
    red[tid] = sum;
    __syncthreads();
    for (int s = 128; s > 0; s >>= 1) {
        if (tid < s) red[tid] += red[tid + s];
        __syncthreads();
    }
    const float inv = 1.0f / red[0];
    float4* o4 = (float4*)out_row;
    const float4* v4 = (const float4*)vals;
    for (int i = tid; i < L_COLS / 4; i += 256) {
        float4 v = v4[i];
        v.x *= inv; v.y *= inv; v.z *= inv; v.w *= inv;
        o4[i] = v;
    }
}

// ---------------------------------------------------------------------------
// Kernel 3: in-place softmax over rows of the user-energy slice.
// ---------------------------------------------------------------------------
__global__ __launch_bounds__(256) void softmax_inplace(float* __restrict__ base) {
    __shared__ __attribute__((aligned(16))) float vals[L_COLS];   // 16 KB
    __shared__ float red[256];
    const int tid = threadIdx.x;
    float* row = base + (size_t)blockIdx.x * L_COLS;
    const float4* r4 = (const float4*)row;
    float4* v4 = (float4*)vals;
    for (int i = tid; i < L_COLS / 4; i += 256) v4[i] = r4[i];
    __syncthreads();
    softmax_row_lds(vals, red, tid, row);
}

// ---------------------------------------------------------------------------
// Kernel 4: fused gather + energy + softmax for attn_loc (y=0) / attn_time (y=1)
// ---------------------------------------------------------------------------
__global__ __launch_bounds__(256) void gather_softmax(
    const int* __restrict__ his, const int* __restrict__ cur,
    const int* __restrict__ his_t, const int* __restrict__ cur_t,
    const float* __restrict__ poi_mat, const float* __restrict__ time_mat,
    float* __restrict__ out)  // d_out base: [3][S][L]
{
    __shared__ __attribute__((aligned(16))) float vals[L_COLS];
    __shared__ float red[256];
    const int s = blockIdx.x;
    const int type = blockIdx.y;
    const int tid = threadIdx.x;

    if (type == 0) {
        const float* row = poi_mat + (size_t)cur[s] * N_POI;
        for (int l = tid; l < L_COLS; l += 256) {
            const float d = row[his[l]];
            vals[l] = (d != 0.0f) ? (1.0f / d) : 1e-6f;
        }
    } else {
        const float* row = time_mat + (size_t)cur_t[s] * N_TIME;
        for (int l = tid; l < L_COLS; l += 256) {
            vals[l] = row[his_t[l]];
        }
    }
    __syncthreads();

    float* orow = out + (size_t)type * S_ROWS * L_COLS + (size_t)s * L_COLS;
    softmax_row_lds(vals, red, tid, orow);
}

// ---------------------------------------------------------------------------
extern "C" void kernel_launch(void* const* d_in, const int* in_sizes, int n_in,
                              void* d_out, int out_size, void* d_ws, size_t ws_size,
                              hipStream_t stream) {
    const int*   his   = (const int*)d_in[0];
    const int*   cur   = (const int*)d_in[1];
    const int*   his_t = (const int*)d_in[2];
    const int*   cur_t = (const int*)d_in[3];
    const float* poi   = (const float*)d_in[4];
    const float* tim   = (const float*)d_in[5];
    const float* uemb  = (const float*)d_in[6];
    const float* iemb  = (const float*)d_in[7];
    const float* soc   = (const float*)d_in[8];
    const float* w1    = (const float*)d_in[9];
    const float* w2    = (const float*)d_in[10];

    float* out  = (float*)d_out;
    float* bias = (float*)d_ws;  // 4096 floats of scratch
    float* user_slice = out + (size_t)2 * S_ROWS * L_COLS;

    // 1) bias vector
    bias_kernel<<<L_COLS, 256, 0, stream>>>(iemb, soc, bias);

    // 2) user energy via WMMA bf16 GEMM
    dim3 ggrid(L_COLS / BN, S_ROWS / BM);
    user_energy_wmma<<<ggrid, 256, 0, stream>>>(uemb, iemb, bias, w1, w2, user_slice);

    // 3) softmax user energy in place
    softmax_inplace<<<S_ROWS, 256, 0, stream>>>(user_slice);

    // 4) fused gather + softmax for loc/time slices
    dim3 sgrid(S_ROWS, 2);
    gather_softmax<<<sgrid, 256, 0, stream>>>(his, cur, his_t, cur_t, poi, tim, out);
}
